// GATLayer_19172734010024
// MI455X (gfx1250) — matrix operand order repaired and checked by reading, not compiled, not run on hardware
//
#include <hip/hip_runtime.h>

#define NN 3000
#define NPAD 3008
#define FIN 64
#define HH 4
#define DD 32
#define BB 2

typedef __attribute__((ext_vector_type(16))) _Float16 v16h;
typedef __attribute__((ext_vector_type(8)))  _Float16 v8h;
typedef __attribute__((ext_vector_type(8)))  float    v8f;

// ---------------------------------------------------------------------------
// Kernel 1: feat = h @ kernel (per b,h); write featT (f16, [b,h,d,n] padded to
// NPAD for 16B-aligned WMMA B-operand loads), E2p[m] = exp(feat.att_k2) and
// node mask. Softmax is shift-invariant in e1 (constant per row), and the
// adjacency is binary {0,1}, so attention needs only exp(e2) — computed ONCE
// here (24K exps) instead of per score (72M exps).
// Block = 256 thr = 8 rows x 32 d-lanes (wave32: one n-row per wave).
// ---------------------------------------------------------------------------
__global__ __launch_bounds__(256) void gat_feat_kernel(
    const float* __restrict__ h, const float* __restrict__ kern,
    const float* __restrict__ k2,
    _Float16* __restrict__ featT, float* __restrict__ E2p,
    float* __restrict__ nodeMask)
{
  __shared__ float sk[FIN * DD];
  const int tid  = threadIdx.x;
  const int lane = tid & 31;
  const int hh   = blockIdx.y, b = blockIdx.z;
  const int bh   = b * HH + hh;

  // stage kernel[h] (64x32 f32 = 8KB) into LDS, shared by the 8 rows
  for (int i = tid; i < FIN * DD; i += 256) sk[i] = kern[hh * FIN * DD + i];
  __syncthreads();

  const int n = blockIdx.x * 8 + (tid >> 5);
  if (n < NN) {
    const float* hrow = h + ((size_t)b * NN + n) * FIN;
    float acc = 0.f;
    int   nz  = 0;
    for (int f = 0; f < FIN; ++f) {
      const float hv = hrow[f];             // broadcast across the wave
      nz |= (hv != 0.f);
      acc = fmaf(hv, sk[f * DD + lane], acc);
    }
    featT[((size_t)bh * DD + lane) * NPAD + n] = (_Float16)acc;

    float p2 = acc * k2[hh * DD + lane];
    for (int w = 16; w; w >>= 1)            // wave32 tree reduce (e2 = feat.k2)
      p2 += __shfl_xor(p2, w, 32);
    if (lane == 0) {
      E2p[(size_t)bh * NPAD + n] = __expf(p2);
      if (hh == 0) nodeMask[(size_t)b * NN + n] = nz ? 1.f : 0.f;
    }
  }
  // zero the E2p padding tail once per (b,h): pad columns contribute 0
  if (blockIdx.x == 0 && tid < (NPAD - NN))
    E2p[(size_t)bh * NPAD + NN + tid] = 0.f;
}

// ---------------------------------------------------------------------------
// Kernel 2: single streaming pass. One wave owns a 16-row tile of one (b,h).
// Per 32-wide m-tile: P = (adj != 0) ? E2p[m] : 0 built branchlessly directly
// in the f16 A-matrix VGPR layout, row-sums accumulated in the same loop,
// two v_wmma_f32_16x16x32_f16 (d=0..15 / 16..31). Epilogue: /rowsum, +bias,
// relu, node-mask. Adjacency is read exactly once (72MB; L2-resident for the
// 4 heads sharing each row tile).
// ---------------------------------------------------------------------------
__global__ __launch_bounds__(128) void gat_attn_kernel(
    const float* __restrict__ adj, const float* __restrict__ bias,
    const _Float16* __restrict__ featT, const float* __restrict__ E2p,
    const float* __restrict__ nodeMask, float* __restrict__ out)
{
  const int lane  = threadIdx.x & 31;
  const int TILES = (NN + 15) / 16;
  const int tile  = blockIdx.x * 4 + (threadIdx.x >> 5);
  if (tile >= TILES) return;                 // wave-uniform exit
  const int hh = blockIdx.y, b = blockIdx.z;
  const int bh = b * HH + hh;

  const float*    abase = adj   + (size_t)b  * NN * NN;
  const float*    E2bh  = E2p   + (size_t)bh * NPAD;
  const _Float16* fT    = featT + (size_t)bh * DD * NPAD;

  const int rA  = lane & 15;                 // A-matrix row for this lane
  int nA = tile * 16 + rA; if (nA >= NN) nA = NN - 1;
  const float* arowA = abase + (size_t)nA * NN;
  const int offA = (lane >= 16) ? 8  : 0;    // A layout: K = off..off+7, off+16..off+23
  const int offB = (lane >= 16) ? 8  : 0;    // B layout: K = off..off+7, off+16..off+23
  const int col  = lane & 15;
  const _Float16* B0p = fT + (size_t)col        * NPAD;  // d = col
  const _Float16* B1p = fT + (size_t)(col + 16) * NPAD;  // d = col+16

  v8f O0 = {}; v8f O1 = {};
  float myS = 0.f;                           // partial row-sum (this lane's 16 K's)

  for (int mb = 0; mb < NPAD; mb += 32) {
    v16h A;
    if (mb + 32 <= NN) {                     // fast path: vector loads, branchless
      #pragma unroll
      for (int g = 0; g < 2; ++g) {
        const int m0 = mb + offA + g * 16;
        const float4 a0 = *(const float4*)(arowA + m0);
        const float4 a1 = *(const float4*)(arowA + m0 + 4);
        const float4 q0 = *(const float4*)(E2bh + m0);
        const float4 q1 = *(const float4*)(E2bh + m0 + 4);
        const float av[8] = {a0.x, a0.y, a0.z, a0.w, a1.x, a1.y, a1.z, a1.w};
        const float ev[8] = {q0.x, q0.y, q0.z, q0.w, q1.x, q1.y, q1.z, q1.w};
        #pragma unroll
        for (int j = 0; j < 8; ++j) {
          const float p = (av[j] != 0.f) ? ev[j] : 0.f;   // v_cndmask, no branch
          myS += p;
          A[g * 8 + j] = (_Float16)p;
        }
      }
    } else {                                 // last m-tile: guarded scalar path
      #pragma unroll
      for (int g = 0; g < 2; ++g) {
        const int m0 = mb + offA + g * 16;
        #pragma unroll
        for (int j = 0; j < 8; ++j) {
          const int   m  = m0 + j;
          const float av = (m < NN) ? arowA[m] : 0.f;
          const float ev = E2bh[m];          // padded; pad values are 0
          const float p  = (av != 0.f) ? ev : 0.f;
          myS += p;
          A[g * 8 + j] = (_Float16)p;
        }
      }
    }
    // B operands: 2x contiguous 16B loads per d-half (featT padded to NPAD)
    const v8h b00 = *(const v8h*)(B0p + mb + offB);
    const v8h b01 = *(const v8h*)(B0p + mb + offB + 16);
    const v8h b10 = *(const v8h*)(B1p + mb + offB);
    const v8h b11 = *(const v8h*)(B1p + mb + offB + 16);
    v16h Bv0, Bv1;
    #pragma unroll
    for (int j = 0; j < 8; ++j) {
      Bv0[j] = b00[j]; Bv0[j + 8] = b01[j];
      Bv1[j] = b10[j]; Bv1[j + 8] = b11[j];
    }
    O0 = __builtin_amdgcn_wmma_f32_16x16x32_f16(false, A, false, Bv0, (short)0, O0, false, false);
    O1 = __builtin_amdgcn_wmma_f32_16x16x32_f16(false, A, false, Bv1, (short)0, O1, false, false);
  }

  // combine the two half-row partners (lane r and lane r+16 split each row's K)
  myS += __shfl_xor(myS, 16, 32);

  // ---------------- epilogue: /rowsum, +bias, relu, node-mask ----------------
  const float bs0 = bias[hh * DD + col];
  const float bs1 = bias[hh * DD + col + 16];
  #pragma unroll
  for (int j = 0; j < 8; ++j) {
    const int r = (lane < 16) ? j : (j + 8);   // C/D layout: M = j (+8 for hi lanes)
    const int n = tile * 16 + r;
    const float Ls    = __shfl(myS, r, 32);    // lanes r and r+16 hold row r's sum
    const float denom = (Ls > 0.f) ? Ls : 1.f; // fully-masked row: O==0 anyway
    if (n < NN) {
      const float mk = nodeMask[(size_t)b * NN + n];
      const float o0 = fmaxf(O0[j] / denom + bs0, 0.f) * mk;
      const float o1 = fmaxf(O1[j] / denom + bs1, 0.f) * mk;
      float* orow = out + ((size_t)(b * NN + n) * HH + hh) * DD;
      orow[col]      = o0;
      orow[col + 16] = o1;
    }
  }
}

// ---------------------------------------------------------------------------
extern "C" void kernel_launch(void* const* d_in, const int* in_sizes, int n_in,
                              void* d_out, int out_size, void* d_ws, size_t ws_size,
                              hipStream_t stream) {
  (void)in_sizes; (void)n_in; (void)out_size; (void)ws_size;
  const float* h    = (const float*)d_in[0];
  const float* a    = (const float*)d_in[1];
  const float* kern = (const float*)d_in[2];
  const float* k2   = (const float*)d_in[4];   // att_k2 (att_k1 cancels in softmax)
  const float* bias = (const float*)d_in[5];
  float* out = (float*)d_out;

  char* ws = (char*)d_ws;
  _Float16* featT = (_Float16*)ws;   // B*H*D*NPAD f16 = 1,540,096 B (16B aligned)
  const size_t featBytes = (size_t)BB * HH * DD * NPAD * sizeof(_Float16);
  float* E2p      = (float*)(ws + featBytes);
  float* nodeMask = E2p + (size_t)BB * HH * NPAD;

  dim3 g1((NN + 7) / 8, HH, BB);
  gat_feat_kernel<<<g1, 256, 0, stream>>>(h, kern, k2, featT, E2p, nodeMask);

  const int TILES = (NN + 15) / 16;
  dim3 g2((TILES + 3) / 4, HH, BB);
  gat_attn_kernel<<<g2, 128, 0, stream>>>(a, bias, featT, E2p, nodeMask, out);
}
